// MoE_net_44178033607366
// MI455X (gfx1250) — matrix operand order repaired and checked by reading, not compiled, not run on hardware
//
#include <hip/hip_runtime.h>

#define B_TOK 1024
#define IN_DIM 512
#define H_DIM 1024
#define OUT_DIM 512
#define NE 8
#define NLAYERS 10

typedef __bf16 bf16;
typedef __attribute__((ext_vector_type(16))) __bf16 v16bf;
typedef __attribute__((ext_vector_type(8)))  __bf16 v8bf;
typedef __attribute__((ext_vector_type(8)))  float  v8f;

// ---------------------------------------------------------------------------
// WMMA fragment loaders for v_wmma_f32_16x16x32_bf16 (C/D per CDNA5 ISA 7.12.2; A/B k order is this code's own)
// A 16x32 (MxK): lane r=lane&15 -> row M=r; half=lane>>4:
//   lanes 0-15 hold K=0..15, lanes 16-31 hold K=16..31 (contiguous, matches B)
// B 32x16 (KxN), stored transposed BT[N][K]: lane r -> column N=r;
//   lanes 0-15 hold K=0..15, lanes 16-31 hold K=16..31 (contiguous)
// C/D 16x16: VGPR v -> row M = 8*half + v, col N = r
// ---------------------------------------------------------------------------
__device__ __forceinline__ v16bf load_a_frag(const bf16* __restrict__ arow,
                                             int k0, int half) {
  return *reinterpret_cast<const v16bf*>(arow + k0 + 16 * half);
}

__device__ __forceinline__ v16bf load_b_frag(const bf16* __restrict__ BT,
                                             int n, int K, int k0, int half) {
  return *reinterpret_cast<const v16bf*>(BT + (size_t)n * K + k0 + 16 * half);
}

__device__ __forceinline__ v8f wmma_bf16(v16bf a, v16bf b, v8f c) {
  return __builtin_amdgcn_wmma_f32_16x16x32_bf16(false, a, false, b,
                                                 (short)0, c, false, false);
}

__device__ __forceinline__ void atomAddF(float* p, float v) {
  unsafeAtomicAdd(p, v);  // global_atomic_add_f32 (no return)
}

// ---------------------------------------------------------------------------
// prep: convert all weights/inputs to bf16; pre-transpose B matrices to [N][K]
// ---------------------------------------------------------------------------
__global__ void prep(const float* __restrict__ x, const float* __restrict__ enc_w,
                     const float* __restrict__ dec_w, const float* __restrict__ w1,
                     const float* __restrict__ w2,
                     bf16* __restrict__ xb, bf16* __restrict__ encT,
                     bf16* __restrict__ decT, bf16* __restrict__ w1T,
                     bf16* __restrict__ w2T, float* __restrict__ loss) {
  const size_t nX = (size_t)B_TOK * IN_DIM;
  const size_t nEc = (size_t)IN_DIM * H_DIM;
  const size_t nDc = (size_t)H_DIM * OUT_DIM;
  const size_t nW = (size_t)NE * H_DIM * H_DIM;
  const size_t HH = (size_t)H_DIM * H_DIM;
  const size_t total = nX + nEc + nDc + 2 * nW;
  size_t stride = (size_t)gridDim.x * blockDim.x;
  for (size_t i = (size_t)blockIdx.x * blockDim.x + threadIdx.x; i < total;
       i += stride) {
    if (i < nX) {
      xb[i] = (bf16)x[i];
    } else if (i < nX + nEc) {
      size_t j = i - nX;
      size_t hh = j / IN_DIM, ii = j % IN_DIM;           // encT[h][i] = enc_w[i][h]
      encT[j] = (bf16)enc_w[ii * H_DIM + hh];
    } else if (i < nX + nEc + nDc) {
      size_t j = i - nX - nEc;
      size_t oo = j / H_DIM, hh = j % H_DIM;             // decT[o][h] = dec_w[h][o]
      decT[j] = (bf16)dec_w[hh * OUT_DIM + oo];
    } else if (i < nX + nEc + nDc + nW) {
      size_t j = i - nX - nEc - nDc;
      size_t e = j / HH, rem = j % HH;
      size_t f = rem / H_DIM, h = rem % H_DIM;           // w1T[e][f][h] = w1[e][h][f]
      w1T[j] = (bf16)w1[(e * H_DIM + h) * H_DIM + f];
    } else {
      size_t j = i - nX - nEc - nDc - nW;
      size_t e = j / HH, rem = j % HH;
      size_t o = rem / H_DIM, f = rem % H_DIM;           // w2T[e][o][f] = w2[e][f][o]
      w2T[j] = (bf16)w2[(e * H_DIM + f) * H_DIM + o];
    }
  }
  if (blockIdx.x == 0 && threadIdx.x == 0) loss[0] = 0.f;
}

// ---------------------------------------------------------------------------
// Generic forward GEMM:  C[M,N] = act(A[M,K] @ B + bias)
// B pre-transposed (BT[N][K] bf16).  Optional row gather (sparse dispatch),
// optional per-expert striding via blockIdx.z, bf16 and/or f32 output.
// Block = 256 thr (8 waves); block tile 32x256; wave tile 16x64 (4 WMMA accs).
// ---------------------------------------------------------------------------
__global__ void gemm_fwd(const bf16* __restrict__ A, int ldA,
                         const int* __restrict__ row_idx, int rowStride,
                         const bf16* __restrict__ BTbase, size_t strideB,
                         const float* __restrict__ biasBase, int strideBias,
                         bf16* __restrict__ CbBase, size_t strideCb,
                         float* __restrict__ Cf,
                         const int* __restrict__ mcount,
                         int M, int N, int K, int doRelu) {
  int e = blockIdx.z;
  const bf16* BT = BTbase + (size_t)e * strideB;
  const float* bias = biasBase ? biasBase + (size_t)e * strideBias : nullptr;
  bf16* Cb = CbBase ? CbBase + (size_t)e * strideCb : nullptr;
  const int* ridx = row_idx ? row_idx + (size_t)e * rowStride : nullptr;
  int Me = mcount ? mcount[e] : M;

  int tid = threadIdx.x;
  int wave = tid >> 5, lane = tid & 31, r = lane & 15, half = lane >> 4;
  int tile_m = blockIdx.y * 32 + (wave >> 2) * 16;
  if (tile_m >= Me) return;  // wave-uniform exit: EXEC stays all-ones for WMMA
  int tile_n = blockIdx.x * 256 + (wave & 3) * 64;

  int row = tile_m + r;
  int src = ridx ? ridx[row] : row;
  if (src < 0) src = 0;  // padded slot: compute junk, discarded downstream
  const bf16* arow = A + (size_t)src * ldA;

  v8f acc[4] = {{}, {}, {}, {}};
  for (int k0 = 0; k0 < K; k0 += 32) {
    v16bf a = load_a_frag(arow, k0, half);
#pragma unroll
    for (int j = 0; j < 4; ++j) {
      v16bf b = load_b_frag(BT, tile_n + j * 16 + r, K, k0, half);
      acc[j] = wmma_bf16(a, b, acc[j]);
    }
  }
#pragma unroll
  for (int j = 0; j < 4; ++j) {
    int n = tile_n + j * 16 + r;
    float bv = bias ? bias[n] : 0.f;
#pragma unroll
    for (int v = 0; v < 8; ++v) {
      int m = tile_m + half * 8 + v;
      float xv = acc[j][v] + bv;
      if (doRelu) xv = fmaxf(xv, 0.f);
      size_t ci = (size_t)m * N + n;
      if (Cb) Cb[ci] = (bf16)xv;
      if (Cf) Cf[ci] = xv;
    }
  }
}

// ---------------------------------------------------------------------------
// Expert GEMM2 + weighted scatter: out[tok] += gate * (hid_e @ w2_e + b2_e)
// Exactly 2 commutative f32-atomic addends per output element -> deterministic.
// ---------------------------------------------------------------------------
__global__ void gemm_moe2(const bf16* __restrict__ hidBase,
                          const bf16* __restrict__ w2TBase,
                          const float* __restrict__ b2Base,
                          const int* __restrict__ tlistBase,
                          const float* __restrict__ glistBase,
                          const int* __restrict__ pcount,
                          float* __restrict__ out, int N, int K) {
  int e = blockIdx.z;
  const bf16* A = hidBase + (size_t)e * B_TOK * K;
  const bf16* BT = w2TBase + (size_t)e * K * N;
  const float* bias = b2Base + (size_t)e * N;
  const int* tl = tlistBase + (size_t)e * B_TOK;
  const float* gl = glistBase + (size_t)e * B_TOK;
  int Me = pcount[e];

  int tid = threadIdx.x;
  int wave = tid >> 5, lane = tid & 31, r = lane & 15, half = lane >> 4;
  int tile_m = blockIdx.y * 32 + (wave >> 2) * 16;
  if (tile_m >= Me) return;
  int tile_n = blockIdx.x * 256 + (wave & 3) * 64;

  const bf16* arow = A + (size_t)(tile_m + r) * K;
  v8f acc[4] = {{}, {}, {}, {}};
  for (int k0 = 0; k0 < K; k0 += 32) {
    v16bf a = load_a_frag(arow, k0, half);
#pragma unroll
    for (int j = 0; j < 4; ++j) {
      v16bf b = load_b_frag(BT, tile_n + j * 16 + r, K, k0, half);
      acc[j] = wmma_bf16(a, b, acc[j]);
    }
  }
  float bv[4];
#pragma unroll
  for (int j = 0; j < 4; ++j) bv[j] = bias[tile_n + j * 16 + r];
#pragma unroll
  for (int v = 0; v < 8; ++v) {
    int m = tile_m + half * 8 + v;
    int tok = tl[m];
    if (tok < 0) continue;  // padded row
    float g = gl[m];
#pragma unroll
    for (int j = 0; j < 4; ++j) {
      int n = tile_n + j * 16 + r;
      atomAddF(&out[(size_t)tok * N + n], g * (acc[j][v] + bv[j]));
    }
  }
}

// ---------------------------------------------------------------------------
// Gating: one wave per token. logits = h @ w_gate [B,E]; top-2; softmax.
// ---------------------------------------------------------------------------
__global__ void gating(const bf16* __restrict__ hb, const float* __restrict__ w_gate,
                       float* __restrict__ gden, int* __restrict__ tok_e,
                       float* __restrict__ tok_g) {
  int t = blockIdx.x * 8 + (threadIdx.x >> 5);
  int lane = threadIdx.x & 31;
  if (t >= B_TOK) return;
  float acc[NE];
#pragma unroll
  for (int e = 0; e < NE; ++e) acc[e] = 0.f;
  const bf16* hrow = hb + (size_t)t * H_DIM;
  for (int k = lane; k < H_DIM; k += 32) {
    float hv = (float)hrow[k];
    const float* wr = w_gate + (size_t)k * NE;
#pragma unroll
    for (int e = 0; e < NE; ++e) acc[e] += hv * wr[e];
  }
#pragma unroll
  for (int off = 16; off > 0; off >>= 1) {
#pragma unroll
    for (int e = 0; e < NE; ++e) acc[e] += __shfl_xor(acc[e], off, 32);
  }
  if (lane == 0) {
    int e0 = 0;
    float l0 = acc[0];
    for (int e = 1; e < NE; ++e)
      if (acc[e] > l0) { l0 = acc[e]; e0 = e; }   // strict > => lowest idx on ties
    int e1 = -1;
    float l1 = -3.4e38f;
    for (int e = 0; e < NE; ++e)
      if (e != e0 && acc[e] > l1) { l1 = acc[e]; e1 = e; }
    float xr = __expf(l1 - l0);                    // <= 1, no overflow
    float g0 = 1.f / (1.f + xr);
    float g1 = xr * g0;
#pragma unroll
    for (int e = 0; e < NE; ++e)
      gden[(size_t)t * NE + e] = (e == e0) ? g0 : ((e == e1) ? g1 : 0.f);
    tok_e[2 * t] = e0;     tok_g[2 * t] = g0;
    tok_e[2 * t + 1] = e1; tok_g[2 * t + 1] = g1;
  }
}

__global__ void layer_init(float* __restrict__ out, int n, int* __restrict__ cnt) {
  int stride = gridDim.x * blockDim.x;
  for (int i = blockIdx.x * blockDim.x + threadIdx.x; i < n; i += stride)
    out[i] = 0.f;
  int t = blockIdx.x * blockDim.x + threadIdx.x;
  if (t < NE) cnt[t] = 0;
}

__global__ void dispatch(const int* __restrict__ tok_e, const float* __restrict__ tok_g,
                         int* __restrict__ cnt, int* __restrict__ tlist,
                         float* __restrict__ glist) {
  int i = blockIdx.x * blockDim.x + threadIdx.x;
  if (i >= 2 * B_TOK) return;
  int e = tok_e[i];
  int pos = atomicAdd(&cnt[e], 1);
  tlist[e * B_TOK + pos] = i >> 1;
  glist[e * B_TOK + pos] = tok_g[i];
}

__global__ void pad_lists(const int* __restrict__ cnt, int* __restrict__ tlist,
                          float* __restrict__ glist, int* __restrict__ pcnt) {
  int i = blockIdx.x * blockDim.x + threadIdx.x;
  if (i >= NE * B_TOK) return;
  int e = i >> 10, s = i & (B_TOK - 1);
  int c = cnt[e];
  if (s >= c) { tlist[i] = -1; glist[i] = 0.f; }
  if (s == 0) {
    int pc = (c + 31) & ~31;
    pcnt[e] = pc > B_TOK ? B_TOK : pc;
  }
}

// Deterministic importance reduction (fixed tree) + aux loss accumulation.
__global__ void aux_loss(const float* __restrict__ gden, const int* __restrict__ cnt,
                         float* __restrict__ loss) {
  __shared__ float imp[NE];
  int e = threadIdx.x >> 5, lane = threadIdx.x & 31;
  float s = 0.f;
  for (int t = lane; t < B_TOK; t += 32) s += gden[(size_t)t * NE + e];
#pragma unroll
  for (int off = 16; off > 0; off >>= 1) s += __shfl_xor(s, off, 32);
  if (lane == 0) imp[e] = s;
  __syncthreads();
  if (threadIdx.x == 0) {
    float mi = 0.f, ml = 0.f;
    for (int k = 0; k < NE; ++k) { mi += imp[k]; ml += (float)cnt[k]; }
    mi /= NE; ml /= NE;
    float vi = 0.f, vl = 0.f;
    for (int k = 0; k < NE; ++k) {
      float d = imp[k] - mi;       vi += d * d;
      float d2 = (float)cnt[k] - ml; vl += d2 * d2;
    }
    vi /= (NE - 1); vl /= (NE - 1);  // ddof=1
    loss[0] += 0.01f * (vi / (mi * mi + 1e-10f) + vl / (ml * ml + 1e-10f));
  }
}

__global__ void f2b(const float* __restrict__ in, bf16* __restrict__ outp, int n) {
  int stride = gridDim.x * blockDim.x;
  for (int i = blockIdx.x * blockDim.x + threadIdx.x; i < n; i += stride)
    outp[i] = (bf16)in[i];
}

__global__ void write_loss(const float* __restrict__ loss, float* __restrict__ dst) {
  if (threadIdx.x == 0 && blockIdx.x == 0) dst[0] = loss[0];
}

// ---------------------------------------------------------------------------
extern "C" void kernel_launch(void* const* d_in, const int* in_sizes, int n_in,
                              void* d_out, int out_size, void* d_ws, size_t ws_size,
                              hipStream_t stream) {
  (void)in_sizes; (void)n_in; (void)out_size; (void)ws_size;
  const float* x = (const float*)d_in[0];
  const float* enc_w = (const float*)d_in[1];
  const float* enc_b = (const float*)d_in[2];
  const float* w_gate = (const float*)d_in[3];
  const float* w1 = (const float*)d_in[4];
  const float* b1 = (const float*)d_in[5];
  const float* w2 = (const float*)d_in[6];
  const float* b2 = (const float*)d_in[7];
  const float* dec_w = (const float*)d_in[8];
  const float* dec_b = (const float*)d_in[9];
  float* out = (float*)d_out;

  const size_t nX = (size_t)B_TOK * IN_DIM;
  const size_t nEc = (size_t)IN_DIM * H_DIM;
  const size_t nDc = (size_t)H_DIM * OUT_DIM;
  const size_t nW = (size_t)NE * H_DIM * H_DIM;
  const size_t nBH = (size_t)B_TOK * H_DIM;

  unsigned char* p = (unsigned char*)d_ws;
  auto take = [&](size_t bytes) -> void* {
    void* r = (void*)p;
    p += (bytes + 255) & ~(size_t)255;
    return r;
  };
  bf16* xb = (bf16*)take(nX * 2);
  bf16* encT = (bf16*)take(nEc * 2);
  bf16* decT = (bf16*)take(nDc * 2);
  bf16* w1T = (bf16*)take(nW * 2);
  bf16* w2T = (bf16*)take(nW * 2);
  bf16* hb = (bf16*)take(nBH * 2);
  float* hout = (float*)take(nBH * 4);
  bf16* hid = (bf16*)take((size_t)NE * nBH * 2);
  float* gden = (float*)take((size_t)B_TOK * NE * 4);
  int* tok_e = (int*)take(2 * B_TOK * 4);
  float* tok_g = (float*)take(2 * B_TOK * 4);
  int* tlist = (int*)take((size_t)NE * B_TOK * 4);
  float* glist = (float*)take((size_t)NE * B_TOK * 4);
  int* cnt = (int*)take(NE * 4);
  int* pcnt = (int*)take(NE * 4);
  float* lossp = (float*)take(4);

  // 1) convert + transpose all weights to bf16, zero loss
  prep<<<4096, 256, 0, stream>>>(x, enc_w, dec_w, w1, w2, xb, encT, decT, w1T,
                                 w2T, lossp);

  // 2) encoder: hb = relu(x @ enc_w + enc_b)   [1024,512]x[512,1024]
  gemm_fwd<<<dim3(H_DIM / 256, B_TOK / 32, 1), 256, 0, stream>>>(
      xb, IN_DIM, nullptr, 0, encT, 0, enc_b, 0, hb, 0, nullptr, nullptr,
      B_TOK, H_DIM, IN_DIM, 1);

  // 3) 10 MoE layers (shared params), sparse top-2 dispatch
  for (int layer = 0; layer < NLAYERS; ++layer) {
    layer_init<<<1024, 256, 0, stream>>>(hout, (int)nBH, cnt);
    gating<<<B_TOK / 8, 256, 0, stream>>>(hb, w_gate, gden, tok_e, tok_g);
    dispatch<<<(2 * B_TOK) / 256, 256, 0, stream>>>(tok_e, tok_g, cnt, tlist, glist);
    pad_lists<<<(NE * B_TOK) / 256, 256, 0, stream>>>(cnt, tlist, glist, pcnt);
    aux_loss<<<1, 256, 0, stream>>>(gden, cnt, lossp);
    // hid_e = relu(gather(hb) @ w1_e + b1_e)
    gemm_fwd<<<dim3(H_DIM / 256, B_TOK / 32, NE), 256, 0, stream>>>(
        hb, H_DIM, tlist, B_TOK, w1T, (size_t)H_DIM * H_DIM, b1, H_DIM, hid,
        nBH, nullptr, pcnt, B_TOK, H_DIM, H_DIM, 1);
    // hout[tok] += g * (hid_e @ w2_e + b2_e)
    gemm_moe2<<<dim3(H_DIM / 256, B_TOK / 32, NE), 256, 0, stream>>>(
        hid, w2T, b2, tlist, glist, pcnt, hout, H_DIM, H_DIM);
    f2b<<<1024, 256, 0, stream>>>(hout, hb, (int)nBH);
  }

  // 4) decoder: out = relu(hb @ dec_w + dec_b)  -> d_out[0 .. B*OUT)
  gemm_fwd<<<dim3(OUT_DIM / 256, B_TOK / 32, 1), 256, 0, stream>>>(
      hb, H_DIM, nullptr, 0, decT, 0, dec_b, 0, nullptr, 0, out, nullptr,
      B_TOK, OUT_DIM, H_DIM, 1);

  // 5) loss scalar -> d_out[B*OUT]
  write_loss<<<1, 1, 0, stream>>>(lossp, out + (size_t)B_TOK * OUT_DIM);
}
